// A2Nework_5566277615913
// MI455X (gfx1250) — compile-verified, matrix operand order checked by reading, not run on hardware
//
#include <hip/hip_runtime.h>

// ---------------------------------------------------------------------------
// A2N attention block for MI455X (gfx1250, wave32, WMMA bf16 16x16x32).
// out = x + w4 @ ( (A @ softmax_b(Bp)^T) @ softmax_b(V) ) + b4
// refactored as  x + ((w4 @ G) @ att_vecs) + b4  to kill the HW-sized conv4.
// ---------------------------------------------------------------------------

typedef __attribute__((ext_vector_type(16))) __bf16 bvec16;
typedef __attribute__((ext_vector_type(8)))  float  fvec8;

#define BATCH 8
#define C_IN  512
#define CM    512
#define CN    256
#define HW    4096
#define MO    1024   // CM + CN + CN rows of the fused conv123 weight

// ---- workspace layout (bytes) ---------------------------------------------
static constexpr size_t SZ_XT  = (size_t)BATCH * HW * C_IN * 2;  // bf16 xT[b][p][c]
static constexpr size_t SZ_WC  = (size_t)MO * C_IN * 2;          // bf16 [w1;w2;w3]
static constexpr size_t SZ_W4C = (size_t)C_IN * CM * 2;          // bf16 w4
static constexpr size_t SZ_A   = (size_t)BATCH * CM * HW * 2;    // bf16 A[b][m][p]
static constexpr size_t SZ_BP  = (size_t)BATCH * CN * HW * 4;    // f32 Bp
static constexpr size_t SZ_V   = SZ_BP;                          // f32 V
static constexpr size_t SZ_AM  = (size_t)BATCH * CN * HW * 2;    // bf16 att_maps[b][n][p]
static constexpr size_t SZ_AVT = SZ_AM;                          // bf16 att_vecsT[b][p][n]
static constexpr size_t SZ_GT  = (size_t)BATCH * CN * CM * 2;    // bf16 Gt[b][n][m]
// Hb bf16 [b][o][n] after Gt

static constexpr size_t OFF_XT  = 0;
static constexpr size_t OFF_WC  = OFF_XT + SZ_XT;
static constexpr size_t OFF_W4C = OFF_WC + SZ_WC;
static constexpr size_t OFF_A   = OFF_W4C + SZ_W4C;
static constexpr size_t OFF_BP  = OFF_A + SZ_A;
static constexpr size_t OFF_V   = OFF_BP + SZ_BP;
static constexpr size_t OFF_AM  = OFF_V + SZ_V;
static constexpr size_t OFF_AVT = OFF_AM + SZ_AM;
static constexpr size_t OFF_GT  = OFF_AVT + SZ_AVT;
static constexpr size_t OFF_HB  = OFF_GT + SZ_GT;

// ---------------------------------------------------------------------------
// WMMA operand fragment load. Works for the A operand from row-major [M][K]
// and for the B operand when B^T is row-major [N][K] (true for every GEMM
// here by construction). Matches ISA bf16 16x32 layout:
//   lane L: row = L&15, chunk0 = K[(L>>4)*8 .. +7], chunk1 = chunk0 + 16.
// ---------------------------------------------------------------------------
__device__ __forceinline__ bvec16 load_frag(const __bf16* __restrict__ tile,
                                            int ld, int lane) {
  const __bf16* p = tile + (size_t)(lane & 15) * ld + ((lane >> 4) << 3);
  union { uint4 q[2]; bvec16 v; } u;
  u.q[0] = *(const uint4*)(p);
  u.q[1] = *(const uint4*)(p + 16);
  return u.v;
}

// Per-wave 32x64 output tile: 2 (M) x 4 (N) WMMA 16x16 tiles, K step 32.
__device__ __forceinline__ void wave_gemm_2x4(const __bf16* __restrict__ Arows,
                                              int lda, int m0,
                                              const __bf16* __restrict__ Brows,
                                              int ldb, int n0, int K, int lane,
                                              fvec8 acc[2][4]) {
  for (int k = 0; k < K; k += 32) {
    bvec16 af[2], bfr[4];
#pragma unroll
    for (int i = 0; i < 2; ++i)
      af[i] = load_frag(Arows + (size_t)(m0 + 16 * i) * lda + k, lda, lane);
#pragma unroll
    for (int j = 0; j < 4; ++j)
      bfr[j] = load_frag(Brows + (size_t)(n0 + 16 * j) * ldb + k, ldb, lane);
#pragma unroll
    for (int i = 0; i < 2; ++i)
#pragma unroll
      for (int j = 0; j < 4; ++j)
        acc[i][j] = __builtin_amdgcn_wmma_f32_16x16x32_bf16(
            false, af[i], false, bfr[j], (short)0, acc[i][j], false, false);
  }
}

// ---------------------------------------------------------------------------
// Prep 1: x [b][c][p] f32  ->  xT [b][p][c] bf16  (LDS tile transpose)
// ---------------------------------------------------------------------------
__global__ __launch_bounds__(256) void a2n_transpose_x(
    const float* __restrict__ x, __bf16* __restrict__ xT) {
  __shared__ float tile[32][33];
  const int b = blockIdx.z;
  const int cb = blockIdx.y * 32;
  const int pb = blockIdx.x * 32;
  const int col = threadIdx.x & 31;
  const int row8 = threadIdx.x >> 5;
  const float* xb = x + (size_t)b * C_IN * HW;
#pragma unroll
  for (int i = 0; i < 4; ++i) {
    int r = row8 + i * 8;
    tile[r][col] = xb[(size_t)(cb + r) * HW + pb + col];
  }
  __syncthreads();
  __bf16* xTb = xT + (size_t)b * HW * C_IN;
#pragma unroll
  for (int i = 0; i < 4; ++i) {
    int r = row8 + i * 8;
    xTb[(size_t)(pb + r) * C_IN + cb + col] = (__bf16)tile[col][r];
  }
}

// ---------------------------------------------------------------------------
// Prep 2: convert weights to bf16, fuse w1|w2|w3 into Wc [1024][512]
// ---------------------------------------------------------------------------
__global__ __launch_bounds__(256) void a2n_convert_weights(
    const float* __restrict__ w1, const float* __restrict__ w2,
    const float* __restrict__ w3, const float* __restrict__ w4,
    __bf16* __restrict__ Wc, __bf16* __restrict__ W4c) {
  const int t = blockIdx.x * 256 + threadIdx.x;
  const int n1 = MO * C_IN;          // 524288
  const int n2 = n1 + C_IN * CM;     // + 262144
  if (t < n1) {
    const int row = t / C_IN;
    float v;
    if (row < CM)            v = w1[t];
    else if (row < CM + CN)  v = w2[t - CM * C_IN];
    else                     v = w3[t - (CM + CN) * C_IN];
    Wc[t] = (__bf16)v;
  } else if (t < n2) {
    W4c[t - n1] = (__bf16)w4[t - n1];
  }
}

// ---------------------------------------------------------------------------
// K1: fused conv1/2/3.  Y[o][p] = sum_c Wc[o][c] * xT[p][c]  (+bias)
// rows 0..511 -> A (bf16), 512..767 -> Bp (f32), 768..1023 -> V (f32)
// ---------------------------------------------------------------------------
__global__ __launch_bounds__(256) void a2n_conv123(
    const __bf16* __restrict__ Wc, const __bf16* __restrict__ xT,
    const float* __restrict__ b1, const float* __restrict__ b2,
    const float* __restrict__ b3, __bf16* __restrict__ Aout,
    float* __restrict__ Bp, float* __restrict__ V) {
  const int b = blockIdx.z;
  const int lane = threadIdx.x & 31;
  const int wave = threadIdx.x >> 5;
  const int m0 = blockIdx.y * 128 + (wave >> 1) * 32;
  const int n0 = blockIdx.x * 128 + (wave & 1) * 64;
  const __bf16* Brows = xT + (size_t)b * HW * C_IN;

  fvec8 zero = {};
  fvec8 acc[2][4];
#pragma unroll
  for (int i = 0; i < 2; ++i)
#pragma unroll
    for (int j = 0; j < 4; ++j) acc[i][j] = zero;

  wave_gemm_2x4(Wc, C_IN, m0, Brows, C_IN, n0, C_IN, lane, acc);

#pragma unroll
  for (int i = 0; i < 2; ++i)
#pragma unroll
    for (int j = 0; j < 4; ++j)
#pragma unroll
      for (int e = 0; e < 8; ++e) {
        const int mg = m0 + 16 * i + 8 * (lane >> 4) + e;
        const int p = n0 + 16 * j + (lane & 15);
        float bias = (mg < CM) ? b1[mg]
                   : (mg < CM + CN) ? b2[mg - CM] : b3[mg - CM - CN];
        const float val = acc[i][j][e] + bias;
        if (mg < CM)
          Aout[(size_t)b * CM * HW + (size_t)mg * HW + p] = (__bf16)val;
        else if (mg < CM + CN)
          Bp[(size_t)b * CN * HW + (size_t)(mg - CM) * HW + p] = val;
        else
          V[(size_t)b * CN * HW + (size_t)(mg - CM - CN) * HW + p] = val;
      }
}

// ---------------------------------------------------------------------------
// K2: softmax over BATCH axis for Bp and V.
// att_maps written [b][n][p]; att_vecs written transposed [b][p][n].
// ---------------------------------------------------------------------------
__global__ __launch_bounds__(256) void a2n_batch_softmax(
    const float* __restrict__ Bp, const float* __restrict__ V,
    __bf16* __restrict__ attm, __bf16* __restrict__ attvT) {
  const size_t NP = (size_t)CN * HW;
  size_t t = (size_t)blockIdx.x * 256 + threadIdx.x;
  if (t >= 2 * NP) return;
  const bool second = t >= NP;
  const size_t j = second ? t - NP : t;
  const float* src = second ? V : Bp;

  float v[BATCH];
  float mx = -3.4e38f;
#pragma unroll
  for (int b = 0; b < BATCH; ++b) {
    v[b] = src[b * NP + j];
    mx = fmaxf(mx, v[b]);
  }
  float s = 0.f;
#pragma unroll
  for (int b = 0; b < BATCH; ++b) {
    v[b] = __expf(v[b] - mx);
    s += v[b];
  }
  const float inv = 1.f / s;
  if (!second) {
#pragma unroll
    for (int b = 0; b < BATCH; ++b) attm[b * NP + j] = (__bf16)(v[b] * inv);
  } else {
    const int n = (int)(j / HW);
    const int p = (int)(j % HW);
#pragma unroll
    for (int b = 0; b < BATCH; ++b)
      attvT[b * NP + (size_t)p * CN + n] = (__bf16)(v[b] * inv);
  }
}

// ---------------------------------------------------------------------------
// K3: gathered.  G[m][n] = sum_p A[m][p] * attm[n][p], stored transposed
// as Gt[b][n][m] (bf16) so it is a ready-made B^T operand for K4.
// ---------------------------------------------------------------------------
__global__ __launch_bounds__(256) void a2n_gathered(
    const __bf16* __restrict__ A, const __bf16* __restrict__ attm,
    __bf16* __restrict__ Gt) {
  const int b = blockIdx.z;
  const int lane = threadIdx.x & 31;
  const int wave = threadIdx.x >> 5;
  const int m0 = blockIdx.y * 128 + (wave >> 1) * 32;  // over CM
  const int n0 = blockIdx.x * 128 + (wave & 1) * 64;   // over CN

  fvec8 zero = {};
  fvec8 acc[2][4];
#pragma unroll
  for (int i = 0; i < 2; ++i)
#pragma unroll
    for (int j = 0; j < 4; ++j) acc[i][j] = zero;

  wave_gemm_2x4(A + (size_t)b * CM * HW, HW, m0,
                attm + (size_t)b * CN * HW, HW, n0, HW, lane, acc);

  __bf16* Gb = Gt + (size_t)b * CN * CM;
#pragma unroll
  for (int i = 0; i < 2; ++i)
#pragma unroll
    for (int j = 0; j < 4; ++j)
#pragma unroll
      for (int e = 0; e < 8; ++e) {
        const int m = m0 + 16 * i + 8 * (lane >> 4) + e;
        const int n = n0 + 16 * j + (lane & 15);
        Gb[(size_t)n * CM + m] = (__bf16)acc[i][j][e];
      }
}

// ---------------------------------------------------------------------------
// K4: Hm = w4 @ G.  Hm[o][n] = sum_m w4[o][m] * Gt[n][m].  Stored [b][o][n].
// ---------------------------------------------------------------------------
__global__ __launch_bounds__(256) void a2n_w4g(
    const __bf16* __restrict__ W4c, const __bf16* __restrict__ Gt,
    __bf16* __restrict__ Hb) {
  const int b = blockIdx.z;
  const int lane = threadIdx.x & 31;
  const int wave = threadIdx.x >> 5;
  const int m0 = blockIdx.y * 128 + (wave >> 1) * 32;  // over C (out chans)
  const int n0 = blockIdx.x * 128 + (wave & 1) * 64;   // over CN

  fvec8 zero = {};
  fvec8 acc[2][4];
#pragma unroll
  for (int i = 0; i < 2; ++i)
#pragma unroll
    for (int j = 0; j < 4; ++j) acc[i][j] = zero;

  wave_gemm_2x4(W4c, CM, m0, Gt + (size_t)b * CN * CM, CM, n0, CM, lane, acc);

  __bf16* Hbb = Hb + (size_t)b * C_IN * CN;
#pragma unroll
  for (int i = 0; i < 2; ++i)
#pragma unroll
    for (int j = 0; j < 4; ++j)
#pragma unroll
      for (int e = 0; e < 8; ++e) {
        const int o = m0 + 16 * i + 8 * (lane >> 4) + e;
        const int n = n0 + 16 * j + (lane & 15);
        Hbb[(size_t)o * CN + n] = (__bf16)acc[i][j][e];
      }
}

// ---------------------------------------------------------------------------
// K5: out[o][p] = x[o][p] + b4[o] + sum_n Hb[o][n] * attvT[p][n]
// ---------------------------------------------------------------------------
__global__ __launch_bounds__(256) void a2n_distribute_out(
    const __bf16* __restrict__ Hb, const __bf16* __restrict__ attvT,
    const float* __restrict__ b4, const float* __restrict__ x,
    float* __restrict__ out) {
  const int b = blockIdx.z;
  const int lane = threadIdx.x & 31;
  const int wave = threadIdx.x >> 5;
  const int m0 = blockIdx.y * 128 + (wave >> 1) * 32;  // over C
  const int n0 = blockIdx.x * 128 + (wave & 1) * 64;   // over HW

  fvec8 zero = {};
  fvec8 acc[2][4];
#pragma unroll
  for (int i = 0; i < 2; ++i)
#pragma unroll
    for (int j = 0; j < 4; ++j) acc[i][j] = zero;

  wave_gemm_2x4(Hb + (size_t)b * C_IN * CN, CN, m0,
                attvT + (size_t)b * HW * CN, CN, n0, CN, lane, acc);

  const float* xb = x + (size_t)b * C_IN * HW;
  float* ob = out + (size_t)b * C_IN * HW;
#pragma unroll
  for (int i = 0; i < 2; ++i)
#pragma unroll
    for (int j = 0; j < 4; ++j)
#pragma unroll
      for (int e = 0; e < 8; ++e) {
        const int o = m0 + 16 * i + 8 * (lane >> 4) + e;
        const int p = n0 + 16 * j + (lane & 15);
        const size_t idx = (size_t)o * HW + p;
        ob[idx] = acc[i][j][e] + b4[o] + xb[idx];
      }
}

// ---------------------------------------------------------------------------
extern "C" void kernel_launch(void* const* d_in, const int* in_sizes, int n_in,
                              void* d_out, int out_size, void* d_ws,
                              size_t ws_size, hipStream_t stream) {
  const float* x  = (const float*)d_in[0];
  const float* w1 = (const float*)d_in[1];
  const float* b1 = (const float*)d_in[2];
  const float* w2 = (const float*)d_in[3];
  const float* b2 = (const float*)d_in[4];
  const float* w3 = (const float*)d_in[5];
  const float* b3 = (const float*)d_in[6];
  const float* w4 = (const float*)d_in[7];
  const float* b4 = (const float*)d_in[8];
  float* out = (float*)d_out;

  char* ws = (char*)d_ws;
  __bf16* XT   = (__bf16*)(ws + OFF_XT);
  __bf16* WC   = (__bf16*)(ws + OFF_WC);
  __bf16* W4C  = (__bf16*)(ws + OFF_W4C);
  __bf16* Abuf = (__bf16*)(ws + OFF_A);
  float*  BP   = (float*)(ws + OFF_BP);
  float*  Vb   = (float*)(ws + OFF_V);
  __bf16* AM   = (__bf16*)(ws + OFF_AM);
  __bf16* AVT  = (__bf16*)(ws + OFF_AVT);
  __bf16* GT   = (__bf16*)(ws + OFF_GT);
  __bf16* HB   = (__bf16*)(ws + OFF_HB);

  const dim3 blk(256);

  a2n_transpose_x<<<dim3(HW / 32, C_IN / 32, BATCH), blk, 0, stream>>>(x, XT);

  a2n_convert_weights<<<dim3((MO * C_IN + C_IN * CM + 255) / 256), blk, 0,
                        stream>>>(w1, w2, w3, w4, WC, W4C);

  a2n_conv123<<<dim3(HW / 128, MO / 128, BATCH), blk, 0, stream>>>(
      WC, XT, b1, b2, b3, Abuf, BP, Vb);

  a2n_batch_softmax<<<dim3((2 * CN * HW + 255) / 256), blk, 0, stream>>>(
      BP, Vb, AM, AVT);

  a2n_gathered<<<dim3(CN / 128, CM / 128, BATCH), blk, 0, stream>>>(Abuf, AM,
                                                                    GT);

  a2n_w4g<<<dim3(CN / 128, C_IN / 128, BATCH), blk, 0, stream>>>(W4C, GT, HB);

  a2n_distribute_out<<<dim3(HW / 128, C_IN / 128, BATCH), blk, 0, stream>>>(
      HB, AVT, b4, x, out);
}
